// FasterRCNN_10213432230167
// MI455X (gfx1250) — matrix operand order repaired and checked by reading, not compiled, not run on hardware
//
#include <hip/hip_runtime.h>
#include <math.h>

// ---------------------------------------------------------------------------
// Types for CDNA5 WMMA (wave32): D(16x16 f32) = A(16x32 f16) * B(32x16 f16) + C
// ---------------------------------------------------------------------------
typedef _Float16 half_t;
typedef __attribute__((ext_vector_type(16))) _Float16 v16h;
typedef __attribute__((ext_vector_type(8)))  _Float16 v8h;
typedef __attribute__((ext_vector_type(4)))  _Float16 v4h;
typedef __attribute__((ext_vector_type(8)))  float    v8f;
typedef __attribute__((ext_vector_type(4)))  float    v4f;

// LDS row stride in halves: 40 halves = 80B -> b128 accesses land on distinct
// bank groups for 16 consecutive rows/cols (gcd(20,64)=4 -> 16-cycle), and
// every 8-half run is 16B aligned.
#define SA 40
#define SB 40

// Problem constants (match reference)
#define IMG      512
#define TOT_ANC  196416
#define PRE_NMS  1000
#define POST_NMS 1000

// ---------------------------------------------------------------------------
// Fragment loaders (wave32 layouts per CDNA5 ISA 7.12.2), vectorized:
//  A 16x32 f16 row-major in LDS: lane holds row M=lane&15; K-halves
//    [hiA..hiA+7] and [16+hiA..16+hiA+7] with hiA = (lane&16)?8:0
//    -> two ds_load_b128.
//  B 32xN f16 stored COLUMN-major in LDS: lane holds col N=col0+(lane&15);
//    K-halves [hiB..hiB+7] and [16+hiB..16+hiB+7] with hiB = (lane&16)?8:0
//    -> two ds_load_b128.
// ---------------------------------------------------------------------------
__device__ __forceinline__ v16h frag_a(const half_t* As, int row, int lane) {
    const int hiA = (lane & 16) ? 8 : 0;
    v8h a0 = *(const v8h*)&As[row * SA + hiA];
    v8h a1 = *(const v8h*)&As[row * SA + 16 + hiA];
    v16h a;
#pragma unroll
    for (int e = 0; e < 8; ++e) { a[e] = a0[e]; a[8 + e] = a1[e]; }
    return a;
}

__device__ __forceinline__ v16h frag_b(const half_t* Bs, int col, int lane) {
    const int hiB = (lane & 16) ? 8 : 0;
    v8h t0 = *(const v8h*)&Bs[col * SB + hiB];
    v8h t1 = *(const v8h*)&Bs[col * SB + 16 + hiB];
    v16h b;
#pragma unroll
    for (int e = 0; e < 8; ++e) { b[e] = t0[e]; b[8 + e] = t1[e]; }
    return b;
}

// ---------------------------------------------------------------------------
// Generic WMMA GEMM:  C[M,N] = A[M,K] * Bw[K,N] + bias (optional ReLU).
// Requirements: K % 32 == 0, lda % 4 == 0, bias != nullptr (true everywhere).
// Double-buffered LDS: one barrier per K-step; next step's global loads are
// issued while the current step's fragments/WMMAs execute. All staging
// addresses are strength-reduced to running-pointer adds.
// Block: 256 threads = 8 waves; block tile 64x64; each wave computes a 16x32
// slab (one A fragment, two WMMAs per K-step).
// Epilogue "batch-split" store (log2_rpb = log2(rows_per_batch), or 30 for
// a plain [M,N] store with c_batch_stride == 0):
//   b = row >> log2_rpb ; pr = row & ((1<<log2_rpb)-1)
//   addr = C + b*c_batch_stride + pr*ldc + col
// ---------------------------------------------------------------------------
template <typename AT, bool RELU>
__global__ void __launch_bounds__(256)
wmma_gemm_kernel(const AT* __restrict__ A, int lda,
                 const float* __restrict__ Bw, int ldb,
                 const float* __restrict__ bias,
                 float* __restrict__ C, int ldc,
                 int M, int N, int K,
                 int log2_rpb, long long c_batch_stride)
{
    __shared__ half_t As[2][64 * SA];   // 64 rows x 32 k, row-major
    __shared__ half_t Bs[2][64 * SB];   // 64 cols x 32 k, col-major

    const int tid  = threadIdx.x;
    const int lane = tid & 31;
    const int wave = tid >> 5;
    const int tr   = wave >> 1;            // 0..3 : 16-row slab
    const int tc   = wave & 1;             // 0..1 : 32-col slab
    const int m0   = blockIdx.x * 64;
    const int n0   = blockIdx.y * 64;

    // staging coordinates (loop-invariant)
    const int sa_k  = (tid & 7) * 4;       // 4 consecutive k per A vector
    const int sb_n  = tid & 63;            // column owned for B staging
    const int sb_k0 = (tid >> 6) * 8;      // 8 consecutive k per B thread

    int ar_[2]; float aflag[2];
    const AT* aptr[2];
#pragma unroll
    for (int j = 0; j < 2; ++j) {
        int idx = tid + j * 256;
        ar_[j] = idx >> 3;
        int gr = m0 + ar_[j];
        aptr[j] = A + (long long)((gr < M) ? gr : (M - 1)) * lda + sa_k;
        aflag[j] = (gr < M) ? 1.f : 0.f;
    }
    const int   gn    = n0 + sb_n;
    const int   gnc   = (gn < N) ? gn : (N - 1);
    const float bflag = (gn < N) ? 1.f : 0.f;
    const float* bptr = Bw + (long long)sb_k0 * ldb + gnc;   // advances 32*ldb/step

    v4f ra[2]; float rb[8];
    auto load_tile = [&]() {
#pragma unroll
        for (int j = 0; j < 2; ++j) {
            if constexpr (sizeof(AT) == 2) {
                v4h hv = *(const v4h*)aptr[j];
                ra[j][0] = (float)hv[0]; ra[j][1] = (float)hv[1];
                ra[j][2] = (float)hv[2]; ra[j][3] = (float)hv[3];
            } else {
                ra[j] = *(const v4f*)aptr[j];
            }
            aptr[j] += 32;
        }
        const float* q = bptr;
#pragma unroll
        for (int j = 0; j < 8; ++j) { rb[j] = *q; q += ldb; }
        bptr += (long long)32 * ldb;
    };
    auto store_tile = [&](int buf) {
#pragma unroll
        for (int j = 0; j < 2; ++j) {
            v4h hs;
#pragma unroll
            for (int q = 0; q < 4; ++q) hs[q] = (half_t)(ra[j][q] * aflag[j]);
            *(v4h*)&As[buf][ar_[j] * SA + sa_k] = hs;
        }
        v8h hv;
#pragma unroll
        for (int j = 0; j < 8; ++j) hv[j] = (half_t)(rb[j] * bflag);
        *(v8h*)&Bs[buf][sb_n * SB + sb_k0] = hv;
    };

    v8f acc0 = {}, acc1 = {};
    load_tile();
    int buf = 0;
    for (int k0 = 0; k0 < K; k0 += 32) {
        store_tile(buf);
        __syncthreads();
        if (k0 + 32 < K) load_tile();                 // overlap with WMMA below
        if (k0 + 64 < K)
            __builtin_prefetch((const void*)aptr[0], 0, 1);

        v16h a  = frag_a(&As[buf][0], tr * 16 + (lane & 15), lane);
        v16h b0 = frag_b(&Bs[buf][0], tc * 32 + (lane & 15), lane);
        v16h b1 = frag_b(&Bs[buf][0], tc * 32 + 16 + (lane & 15), lane);
        acc0 = __builtin_amdgcn_wmma_f32_16x16x32_f16(false, a, false, b0,
                                                      (short)0, acc0, false, false);
        acc1 = __builtin_amdgcn_wmma_f32_16x16x32_f16(false, a, false, b1,
                                                      (short)0, acc1, false, false);
        buf ^= 1;
    }

    const int rbase = m0 + tr * 16 + ((lane & 16) ? 8 : 0);
    const int rmask = (1 << log2_rpb) - 1;
#pragma unroll
    for (int hf = 0; hf < 2; ++hf) {
        const v8f& acc = hf ? acc1 : acc0;
        int col = n0 + tc * 32 + hf * 16 + (lane & 15);
        float bs = (col < N) ? bias[col] : 0.f;
#pragma unroll
        for (int r = 0; r < 8; ++r) {
            int row = rbase + r;
            if (row < M && col < N) {
                float v = acc[r] + bs;
                if (RELU) v = fmaxf(v, 0.f);
                int bb = row >> log2_rpb;
                int pr = row & rmask;
                C[bb * c_batch_stride + (long long)pr * ldc + col] = v;
            }
        }
    }
}

// ---------------------------------------------------------------------------
// 3x3 SAME conv via implicit-im2col WMMA GEMM. Specialized to Cin=Cout=256,
// square power-of-two maps (all call sites).
// in : [B,W,W,256] f32 ; wgt : [3,3,256,256] HWIO ; out : [B,W,W,256]
// Grid: (W*W/64, 4, B), block 256. Double-buffered like the GEMM; 72 K-steps
// (9 taps x 8 chunks of 32), chunks never straddle a tap. The weight stream
// is a running pointer (advances 32*256/step).
// ---------------------------------------------------------------------------
template <bool RELU>
__global__ void __launch_bounds__(256)
conv3x3_wmma_kernel(const float* __restrict__ in, int LW,
                    const float* __restrict__ wgt,
                    const float* __restrict__ bias,
                    float* __restrict__ out)
{
    __shared__ half_t As[2][64 * SA];
    __shared__ half_t Bs[2][64 * SB];

    const int W    = 1 << LW;
    const int HW   = 1 << (2 * LW);
    const int tid  = threadIdx.x;
    const int lane = tid & 31;
    const int wave = tid >> 5;
    const int tr   = wave >> 1;
    const int tc   = wave & 1;
    const int m0   = blockIdx.x * 64;
    const int n0   = blockIdx.y * 64;
    const int b    = blockIdx.z;
    const float* inb = in + (long long)b * HW * 256;

    const int sa_k  = (tid & 7) * 4;
    const int sb_n  = tid & 63;
    const int sb_k0 = (tid >> 6) * 8;

    int py_[2], px_[2];
#pragma unroll
    for (int j = 0; j < 2; ++j) {
        int idx = tid + j * 256;
        int p = m0 + (idx >> 3);            // p < HW guaranteed
        py_[j] = p >> LW;
        px_[j] = p & (W - 1);
    }
    const float* wptr = wgt + (long long)sb_k0 * 256 + (n0 + sb_n);

    v4f ra[2]; float aflag[2]; float rb[8];
    auto load_tile = [&](int s) {
        int tap = s >> 3;                   // 8 chunks of 32 per tap
        int c0  = (s & 7) << 5;
        int ky  = tap / 3 - 1, kx = tap % 3 - 1;
#pragma unroll
        for (int j = 0; j < 2; ++j) {
            int y = py_[j] + ky, x = px_[j] + kx;
            bool ok = (y >= 0 && y < W && x >= 0 && x < W);
            int yc = min(max(y, 0), W - 1);
            int xc = min(max(x, 0), W - 1);
            aflag[j] = ok ? 1.f : 0.f;
            ra[j] = *(const v4f*)&inb[(((long long)yc << LW) + xc) * 256 + c0 + sa_k];
        }
        const float* q = wptr;
#pragma unroll
        for (int j = 0; j < 8; ++j) { rb[j] = *q; q += 256; }
        wptr += 32 * 256;
    };
    auto store_tile = [&](int buf) {
#pragma unroll
        for (int j = 0; j < 2; ++j) {
            int r = (tid + j * 256) >> 3;
            v4h hs;
#pragma unroll
            for (int q = 0; q < 4; ++q) hs[q] = (half_t)(ra[j][q] * aflag[j]);
            *(v4h*)&As[buf][r * SA + sa_k] = hs;
        }
        v8h hv;
#pragma unroll
        for (int j = 0; j < 8; ++j) hv[j] = (half_t)rb[j];
        *(v8h*)&Bs[buf][sb_n * SB + sb_k0] = hv;
    };

    v8f acc0 = {}, acc1 = {};
    load_tile(0);
    int buf = 0;
    for (int s = 0; s < 72; ++s) {          // 9 taps x (256/32) chunks
        store_tile(buf);
        __syncthreads();
        if (s + 1 < 72) load_tile(s + 1);

        v16h a  = frag_a(&As[buf][0], tr * 16 + (lane & 15), lane);
        v16h b0 = frag_b(&Bs[buf][0], tc * 32 + (lane & 15), lane);
        v16h b1 = frag_b(&Bs[buf][0], tc * 32 + 16 + (lane & 15), lane);
        acc0 = __builtin_amdgcn_wmma_f32_16x16x32_f16(false, a, false, b0,
                                                      (short)0, acc0, false, false);
        acc1 = __builtin_amdgcn_wmma_f32_16x16x32_f16(false, a, false, b1,
                                                      (short)0, acc1, false, false);
        buf ^= 1;
    }

    float* outb = out + (long long)b * HW * 256;
    const int rbase = m0 + tr * 16 + ((lane & 16) ? 8 : 0);
#pragma unroll
    for (int hf = 0; hf < 2; ++hf) {
        const v8f& acc = hf ? acc1 : acc0;
        int col = n0 + tc * 32 + hf * 16 + (lane & 15);
        float bs = bias[col];
#pragma unroll
        for (int r = 0; r < 8; ++r) {
            float v = acc[r] + bs;
            if (RELU) v = fmaxf(v, 0.f);
            outb[(long long)(rbase + r) * 256 + col] = v;
        }
    }
}

// ---------------------------------------------------------------------------
// Elementwise helpers
// ---------------------------------------------------------------------------
__global__ void maxpool2_kernel(const float* __restrict__ in, float* __restrict__ out,
                                int Ho, int Wo, int C, int B)
{
    long long i = (long long)blockIdx.x * blockDim.x + threadIdx.x;
    long long tot = (long long)B * Ho * Wo * C;
    if (i >= tot) return;
    int c = (int)(i % C); long long t = i / C;
    int x = (int)(t % Wo); t /= Wo;
    int y = (int)(t % Ho); t /= Ho;
    int b = (int)t;
    int Hi = Ho * 2, Wi = Wo * 2;
    const float* p = in + (((long long)b * Hi + 2 * y) * Wi + 2 * x) * C + c;
    float v0 = p[0], v1 = p[(long long)C];
    float v2 = p[(long long)Wi * C], v3 = p[(long long)Wi * C + C];
    out[i] = fmaxf(fmaxf(v0, v1), fmaxf(v2, v3));
}

// p[b,y,x,c] += pu[b,y/2,x/2,c]
__global__ void upsample_add_kernel(float* __restrict__ p, const float* __restrict__ pu,
                                    int H, int W, int C, int B)
{
    long long i = (long long)blockIdx.x * blockDim.x + threadIdx.x;
    long long tot = (long long)B * H * W * C;
    if (i >= tot) return;
    int c = (int)(i % C); long long t = i / C;
    int x = (int)(t % W); t /= W;
    int y = (int)(t % H); t /= H;
    int b = (int)t;
    int Hu = H >> 1, Wu = W >> 1;
    p[i] += pu[(((long long)b * Hu + (y >> 1)) * Wu + (x >> 1)) * C + c];
}

// ---------------------------------------------------------------------------
// Anchor generation + box decode (anchors computed on the fly, clipped first)
// ---------------------------------------------------------------------------
__global__ void decode_kernel(const float* __restrict__ deltas,
                              float* __restrict__ dec, int B)
{
    const int   offs[6]  = {0, 147456, 184320, 193536, 195840, 196416};
    const int   fss[5]   = {128, 64, 32, 16, 8};
    const float sizes[5] = {16.f, 32.f, 64.f, 128.f, 256.f};

    long long i = (long long)blockIdx.x * blockDim.x + threadIdx.x;
    if (i >= (long long)B * TOT_ANC) return;
    int t = (int)(i % TOT_ANC);

    int lvl = 0;
    while (lvl < 4 && t >= offs[lvl + 1]) lvl++;
    int tl   = t - offs[lvl];
    int fs   = fss[lvl];
    int cell = tl / 9, a = tl % 9;
    int y = cell / fs, x = cell % fs;
    float stride = (float)(1 << (lvl + 2));
    float cy = (y + 0.5f) * stride, cx = (x + 0.5f) * stride;

    int   sci = a / 3, ari = a % 3;
    float scale = exp2f((float)sci / 3.0f);
    float ar    = (ari == 0) ? 0.5f : ((ari == 1) ? 1.0f : 2.0f);
    float sz    = sizes[lvl] * scale;
    float sq    = sqrtf(ar);
    float ah0   = sz / sq, aw0 = sz * sq;

    float y1 = fminf(fmaxf(cy - ah0 * 0.5f, 0.f), (float)IMG);
    float x1 = fminf(fmaxf(cx - aw0 * 0.5f, 0.f), (float)IMG);
    float y2 = fminf(fmaxf(cy + ah0 * 0.5f, 0.f), (float)IMG);
    float x2 = fminf(fmaxf(cx + aw0 * 0.5f, 0.f), (float)IMG);

    float ah = y2 - y1, aw = x2 - x1;
    float acy = y1 + ah * 0.5f, acx = x1 + aw * 0.5f;

    const float* d = deltas + i * 4;
    float ncy = d[0] * ah + acy;
    float ncx = d[1] * aw + acx;
    float nh  = expf(d[2]) * ah;
    float nw  = expf(d[3]) * aw;

    float* o = dec + i * 4;
    o[0] = ncy - nh * 0.5f;
    o[1] = ncx - nw * 0.5f;
    o[2] = ncy + nh * 0.5f;
    o[3] = ncx + nw * 0.5f;
}

// ---------------------------------------------------------------------------
// Top-K (K=1000) per batch via iterated block-wide argmax (scores live in L2).
// Copies scores to a work buffer first so graph replays are deterministic.
// ---------------------------------------------------------------------------
__global__ void __launch_bounds__(1024)
topk_kernel(const float* __restrict__ scores, const float* __restrict__ dec,
            float* __restrict__ work,
            float* __restrict__ sc_out, float* __restrict__ boxes_out)
{
    const int b = blockIdx.x, tid = threadIdx.x;
    const float* s = scores + (long long)b * TOT_ANC;
    float* w = work + (long long)b * TOT_ANC;
    for (int i = tid; i < TOT_ANC; i += 1024) w[i] = s[i];
    __syncthreads();

    __shared__ float rv[1024];
    __shared__ int   ri[1024];

    for (int k = 0; k < PRE_NMS; ++k) {
        float best = -__builtin_inff(); int bi = 0;
        for (int i = tid; i < TOT_ANC; i += 1024) {
            float v = w[i];
            if (v > best) { best = v; bi = i; }
        }
        rv[tid] = best; ri[tid] = bi;
        __syncthreads();
        for (int sft = 512; sft > 0; sft >>= 1) {
            if (tid < sft) {
                if (rv[tid + sft] > rv[tid] ||
                    (rv[tid + sft] == rv[tid] && ri[tid + sft] < ri[tid])) {
                    rv[tid] = rv[tid + sft]; ri[tid] = ri[tid + sft];
                }
            }
            __syncthreads();
        }
        if (tid == 0) {
            int idx = ri[0];
            sc_out[b * PRE_NMS + k] = rv[0];
            const float* db = dec + ((long long)b * TOT_ANC + idx) * 4;
            float* bo = boxes_out + ((long long)b * PRE_NMS + k) * 4;
            bo[0] = db[0]; bo[1] = db[1]; bo[2] = db[2]; bo[3] = db[3];
            w[idx] = -__builtin_inff();
        }
        __syncthreads();
    }
}

// ---------------------------------------------------------------------------
// NMS: sequential over i (score order), parallel over j. Then compact into
// rois[b] (zero-filled, kept boxes in order).
// ---------------------------------------------------------------------------
__global__ void __launch_bounds__(256)
nms_kernel(const float* __restrict__ sc, const float* __restrict__ boxes,
           float* __restrict__ rois)
{
    const int b = blockIdx.x, tid = threadIdx.x;
    __shared__ float y1[PRE_NMS], x1[PRE_NMS], y2[PRE_NMS], x2[PRE_NMS], area[PRE_NMS];
    __shared__ unsigned char keep[PRE_NMS];

    const float* bb = boxes + (long long)b * PRE_NMS * 4;
    for (int i = tid; i < PRE_NMS; i += 256) {
        float a0 = bb[i * 4 + 0], a1 = bb[i * 4 + 1];
        float a2 = bb[i * 4 + 2], a3 = bb[i * 4 + 3];
        y1[i] = a0; x1[i] = a1; y2[i] = a2; x2[i] = a3;
        area[i] = (a2 - a0) * (a3 - a1);
        keep[i] = (sc[b * PRE_NMS + i] > 0.0f) ? 1 : 0;
    }
    __syncthreads();

    for (int i = 0; i < PRE_NMS; ++i) {
        if (keep[i]) {
            float iy1 = y1[i], ix1 = x1[i], iy2 = y2[i], ix2 = x2[i], ai = area[i];
            for (int j = i + 1 + tid; j < PRE_NMS; j += 256) {
                float yy1 = fmaxf(iy1, y1[j]), xx1 = fmaxf(ix1, x1[j]);
                float yy2 = fminf(iy2, y2[j]), xx2 = fminf(ix2, x2[j]);
                float inter = fmaxf(yy2 - yy1, 0.f) * fmaxf(xx2 - xx1, 0.f);
                float iou = inter / (ai + area[j] - inter + 1e-8f);
                if (iou > 0.7f) keep[j] = 0;
            }
        }
        __syncthreads();
    }

    float* r = rois + (long long)b * POST_NMS * 4;
    for (int i = tid; i < POST_NMS * 4; i += 256) r[i] = 0.f;
    __syncthreads();
    if (tid == 0) {
        int slot = 0;
        for (int i = 0; i < PRE_NMS; ++i) {
            if (keep[i]) {
                r[slot * 4 + 0] = y1[i]; r[slot * 4 + 1] = x1[i];
                r[slot * 4 + 2] = y2[i]; r[slot * 4 + 3] = x2[i];
                slot++;
            }
        }
    }
}

// ---------------------------------------------------------------------------
// ROI-align: 14x14 bilinear sample grid -> 2x2 average -> 7x7x256.
// Output stored as f16 (feeds the WMMA fc1 GEMM directly).
// Grid: (R=1000, B), block 256 (one channel per thread).
// ---------------------------------------------------------------------------
__global__ void __launch_bounds__(256)
roi_align_kernel(const float* __restrict__ o2, const float* __restrict__ o3,
                 const float* __restrict__ o4, const float* __restrict__ o5,
                 const float* __restrict__ o6,
                 const float* __restrict__ rois,
                 half_t* __restrict__ feats)
{
    const int r = blockIdx.x, b = blockIdx.y, c = threadIdx.x;
    const float* box = rois + ((long long)b * POST_NMS + r) * 4;
    float b0 = box[0], b1 = box[1], b2 = box[2], b3 = box[3];
    float h = b2 - b0, w = b3 - b1;
    float size = sqrtf(fmaxf(h * w, 1e-8f));
    float lv = floorf(log2f(size / 224.f) + 4.f);
    int lvl = (int)fminf(fmaxf(lv, 2.f), 6.f);

    const float* fm; int fs;
    switch (lvl) {
        case 2:  fm = o2; fs = 128; break;
        case 3:  fm = o3; fs = 64;  break;
        case 4:  fm = o4; fs = 32;  break;
        case 5:  fm = o5; fs = 16;  break;
        default: fm = o6; fs = 8;   break;
    }
    float stride = (float)(1 << lvl);
    fm += (long long)b * fs * fs * 256;
    float fsf = (float)fs;

    half_t* out = feats + ((long long)b * POST_NMS + r) * 49 * 256;
    for (int py = 0; py < 7; ++py) {
        for (int px = 0; px < 7; ++px) {
            float acc = 0.f;
#pragma unroll
            for (int sy = 0; sy < 2; ++sy) {
#pragma unroll
                for (int sx = 0; sx < 2; ++sx) {
                    int gy = py * 2 + sy, gx = px * 2 + sx;
                    float yy = b0 / stride + (gy + 0.5f) * (h / stride) / 14.f - 0.5f;
                    yy = fminf(fmaxf(yy, 0.f), fsf - 1.f);
                    float xx = b1 / stride + (gx + 0.5f) * (w / stride) / 14.f - 0.5f;
                    xx = fminf(fmaxf(xx, 0.f), fsf - 1.f);
                    int y0 = (int)floorf(yy), x0 = (int)floorf(xx);
                    int y1i = min(y0 + 1, fs - 1), x1i = min(x0 + 1, fs - 1);
                    float wy = yy - (float)y0, wx = xx - (float)x0;
                    const float* r00 = fm + ((long long)y0 * fs + x0) * 256;
                    const float* r01 = fm + ((long long)y0 * fs + x1i) * 256;
                    const float* r10 = fm + ((long long)y1i * fs + x0) * 256;
                    const float* r11 = fm + ((long long)y1i * fs + x1i) * 256;
                    acc += (1.f - wy) * ((1.f - wx) * r00[c] + wx * r01[c]) +
                           wy        * ((1.f - wx) * r10[c] + wx * r11[c]);
                }
            }
            out[(py * 7 + px) * 256 + c] = (half_t)(acc * 0.25f);
        }
    }
}

// ---------------------------------------------------------------------------
// Host-side orchestration
// ---------------------------------------------------------------------------
static inline int cdiv(int a, int b) { return (a + b - 1) / b; }

extern "C" void kernel_launch(void* const* d_in, const int* in_sizes, int n_in,
                              void* d_out, int out_size, void* d_ws, size_t ws_size,
                              hipStream_t stream)
{
    (void)in_sizes; (void)n_in; (void)out_size; (void)ws_size;

    // ---- inputs (setup_inputs() order) ----
    const float* c2    = (const float*)d_in[0];   // [2,128,128,256]
    const float* c3    = (const float*)d_in[1];   // [2,64,64,512]
    const float* c4    = (const float*)d_in[2];   // [2,32,32,1024]
    const float* c5    = (const float*)d_in[3];   // [2,16,16,2048]
    const float* w_l[5]  = {(const float*)d_in[4],  (const float*)d_in[8],
                            (const float*)d_in[12], (const float*)d_in[16],
                            (const float*)d_in[20]};
    const float* b_l[5]  = {(const float*)d_in[5],  (const float*)d_in[9],
                            (const float*)d_in[13], (const float*)d_in[17],
                            (const float*)d_in[21]};
    const float* w_o[5]  = {(const float*)d_in[6],  (const float*)d_in[10],
                            (const float*)d_in[14], (const float*)d_in[18],
                            (const float*)d_in[22]};
    const float* b_o[5]  = {(const float*)d_in[7],  (const float*)d_in[11],
                            (const float*)d_in[15], (const float*)d_in[19],
                            (const float*)d_in[23]};
    const float* w_rpn = (const float*)d_in[24];
    const float* b_rpn = (const float*)d_in[25];
    const float* w_obj = (const float*)d_in[26];
    const float* b_obj = (const float*)d_in[27];
    const float* w_dlt = (const float*)d_in[28];
    const float* b_dlt = (const float*)d_in[29];
    const float* w_fc1 = (const float*)d_in[30];
    const float* b_fc1 = (const float*)d_in[31];
    const float* w_fc2 = (const float*)d_in[32];
    const float* b_fc2 = (const float*)d_in[33];
    const float* w_box = (const float*)d_in[34];
    const float* b_box = (const float*)d_in[35];
    const float* w_cls = (const float*)d_in[36];
    const float* b_cls = (const float*)d_in[37];

    const int B = 2;
    const int lw_lvl[5]  = {7, 6, 5, 4, 3};            // log2(feature size)
    const int fs_lvl[5]  = {128, 64, 32, 16, 8};
    const int cin_lvl[5] = {256, 512, 1024, 2048, 2048};
    const int anc_off[5] = {0, 147456, 184320, 193536, 195840};

    // ---- workspace carve-up (float units) ----
    float* ws = (float*)d_ws;
    size_t off = 0;
    auto alloc = [&](size_t n) { float* p = ws + off; off += n; return p; };

    float* c6   = alloc((size_t)B * 8 * 8 * 2048);            // 262144
    float* p[5], * o[5];
    for (int l = 0; l < 5; ++l) p[l] = alloc((size_t)B * fs_lvl[l] * fs_lvl[l] * 256);
    for (int l = 0; l < 5; ++l) o[l] = alloc((size_t)B * fs_lvl[l] * fs_lvl[l] * 256);
    float* tbuf   = alloc((size_t)B * 128 * 128 * 256);       // RPN tower (max level)
    float* scores = alloc((size_t)B * TOT_ANC);
    float* deltas = alloc((size_t)B * TOT_ANC * 4);
    float* dec    = alloc((size_t)B * TOT_ANC * 4);
    float* work   = alloc((size_t)B * TOT_ANC);
    float* sc_top = alloc((size_t)B * PRE_NMS);
    float* bx_top = alloc((size_t)B * PRE_NMS * 4);
    half_t* feats = (half_t*)alloc((size_t)B * POST_NMS * 12544 / 2);
    float* x1buf  = alloc((size_t)B * POST_NMS * 1024);
    float* x2buf  = alloc((size_t)B * POST_NMS * 1024);

    float* out_rois = (float*)d_out;            // [2,1000,4]
    float* out_box  = out_rois + 8000;          // [2,1000,4]
    float* out_cls  = out_rois + 16000;         // [2,1000,81]

    // ---- 1. c6 = maxpool2(c5) ----
    {
        long long tot = (long long)B * 8 * 8 * 2048;
        maxpool2_kernel<<<dim3((unsigned)((tot + 255) / 256)), 256, 0, stream>>>(
            c5, c6, 8, 8, 2048, B);
    }

    // ---- 2. lateral 1x1 convs (GEMM) ----
    const float* lat_in[5] = {c2, c3, c4, c5, c6};
    for (int l = 0; l < 5; ++l) {
        int M = B * fs_lvl[l] * fs_lvl[l];
        int K = cin_lvl[l];
        wmma_gemm_kernel<float, false>
            <<<dim3(cdiv(M, 64), cdiv(256, 64)), 256, 0, stream>>>(
                lat_in[l], K, w_l[l], 256, b_l[l],
                p[l], 256, M, 256, K, 30, 0);
    }

    // ---- 3. top-down: p[l] += upsample2(p[l+1]) (p6 has no add) ----
    for (int l = 3; l >= 0; --l) {
        long long tot = (long long)B * fs_lvl[l] * fs_lvl[l] * 256;
        upsample_add_kernel<<<dim3((unsigned)((tot + 255) / 256)), 256, 0, stream>>>(
            p[l], p[l + 1], fs_lvl[l], fs_lvl[l], 256, B);
    }

    // ---- 4. o[l] = conv3x3(p[l]) ----
    for (int l = 0; l < 5; ++l) {
        int HW = fs_lvl[l] * fs_lvl[l];
        conv3x3_wmma_kernel<false>
            <<<dim3(cdiv(HW, 64), 4, B), 256, 0, stream>>>(
                p[l], lw_lvl[l], w_o[l], b_o[l], o[l]);
    }

    // ---- 5. RPN per level: tower conv3x3+ReLU, then 1x1 score/delta convs
    //         written straight into the per-batch concatenated buffers ----
    for (int l = 0; l < 5; ++l) {
        int HW = fs_lvl[l] * fs_lvl[l];
        int M  = B * HW;
        int lg = 2 * lw_lvl[l];             // log2(HW)
        conv3x3_wmma_kernel<true>
            <<<dim3(cdiv(HW, 64), 4, B), 256, 0, stream>>>(
                o[l], lw_lvl[l], w_rpn, b_rpn, tbuf);
        // scores: [B*HW,256] x [256,9] -> scores + b*TOT + off + p*9 + a
        wmma_gemm_kernel<float, false>
            <<<dim3(cdiv(M, 64), 1), 256, 0, stream>>>(
                tbuf, 256, w_obj, 9, b_obj,
                scores + anc_off[l], 9, M, 9, 256,
                lg, (long long)TOT_ANC);
        // deltas: [B*HW,256] x [256,36] -> deltas + b*TOT*4 + off*4 + p*36
        wmma_gemm_kernel<float, false>
            <<<dim3(cdiv(M, 64), 1), 256, 0, stream>>>(
                tbuf, 256, w_dlt, 36, b_dlt,
                deltas + (long long)anc_off[l] * 4, 36, M, 36, 256,
                lg, (long long)TOT_ANC * 4);
    }

    // ---- 6. decode anchors + deltas ----
    {
        long long tot = (long long)B * TOT_ANC;
        decode_kernel<<<dim3((unsigned)((tot + 255) / 256)), 256, 0, stream>>>(
            deltas, dec, B);
    }

    // ---- 7. top-1000 per batch, 8. NMS -> rois (in d_out) ----
    topk_kernel<<<dim3(B), 1024, 0, stream>>>(scores, dec, work, sc_top, bx_top);
    nms_kernel<<<dim3(B), 256, 0, stream>>>(sc_top, bx_top, out_rois);

    // ---- 9. ROI-align -> feats (f16) ----
    roi_align_kernel<<<dim3(POST_NMS, B), 256, 0, stream>>>(
        o[0], o[1], o[2], o[3], o[4], out_rois, feats);

    // ---- 10. FC head ----
    {
        int M = B * POST_NMS;  // 2000
        wmma_gemm_kernel<half_t, true>
            <<<dim3(cdiv(M, 64), cdiv(1024, 64)), 256, 0, stream>>>(
                feats, 12544, w_fc1, 1024, b_fc1, x1buf, 1024,
                M, 1024, 12544, 30, 0);
        wmma_gemm_kernel<float, true>
            <<<dim3(cdiv(M, 64), cdiv(1024, 64)), 256, 0, stream>>>(
                x1buf, 1024, w_fc2, 1024, b_fc2, x2buf, 1024,
                M, 1024, 1024, 30, 0);
        wmma_gemm_kernel<float, false>
            <<<dim3(cdiv(M, 64), 1), 256, 0, stream>>>(
                x2buf, 1024, w_box, 4, b_box, out_box, 4,
                M, 4, 1024, 30, 0);
        wmma_gemm_kernel<float, false>
            <<<dim3(cdiv(M, 64), cdiv(81, 64)), 256, 0, stream>>>(
                x2buf, 1024, w_cls, 81, b_cls, out_cls, 81,
                M, 81, 1024, 30, 0);
    }
}
